// tied_SplineLinear_FAST_17575006175270
// MI455X (gfx1250) — compile-verified, matrix operand order checked by reading, not run on hardware
//
#include <hip/hip_runtime.h>
#include <hip/hip_bf16.h>
#include <stdint.h>

// out[b,h,o] = sum_{i,d} x[b,h,i,d] * wsum[o,i] * w[o,d]
//   == GEMM:  out[M,N] = X[M,K] * Wt^T   with M=B*H=8192, N=O=1024, K=I*4=4096
//   where Wt[o][i*4+d] = wsum[o,i]*w[o,d]  (stored O x K so each WMMA B lane
//   reads K-contiguous bf16 exactly like the A lanes).

typedef __attribute__((ext_vector_type(16))) __bf16 v16bf;
typedef __attribute__((ext_vector_type(8)))  __bf16 v8bf;
typedef __attribute__((ext_vector_type(4)))  __bf16 v4bf;
typedef __attribute__((ext_vector_type(8)))  float  v8f;
typedef int v4i_gcc __attribute__((vector_size(16)));   // matches builtin params

#define MDIM 8192
#define NDIM 1024
#define KDIM 4096
#define IDIM 1024

#define BM 128
#define BN 128
#define BK 64
#define LDT 72   // padded LDS row stride in bf16 elems (144B, 16B-aligned rows)

#define AS1 __attribute__((address_space(1)))
#define AS3 __attribute__((address_space(3)))

#if defined(__has_builtin)
#if __has_builtin(__builtin_amdgcn_global_load_async_to_lds_b128)
#define USE_ASYNC_LDS 1
#endif
#endif

#if defined(USE_ASYNC_LDS)
__device__ __forceinline__ void async_b128(const __bf16* g, const __bf16* l) {
  __builtin_amdgcn_global_load_async_to_lds_b128(
      (AS1 v4i_gcc*)(uintptr_t)g,
      (AS3 v4i_gcc*)(uint32_t)(uintptr_t)l, 0, 0);
}
__device__ __forceinline__ void wait_async0() {
#if __has_builtin(__builtin_amdgcn_s_wait_asynccnt)
  __builtin_amdgcn_s_wait_asynccnt(0);
#else
  asm volatile("s_wait_asynccnt 0x0" ::: "memory");
#endif
}
#endif

// ---------------- prep 1: x (f32) -> xb (bf16), straight copy-convert ---------
__global__ __launch_bounds__(256) void cvt_x_kernel(const float* __restrict__ x,
                                                    __bf16* __restrict__ xb, int n8) {
  int t = blockIdx.x * 256 + threadIdx.x;
  if (t >= n8) return;
  const float4* p = (const float4*)x + 2 * (size_t)t;
  float4 f0 = p[0], f1 = p[1];
  v8bf o;
  o[0] = (__bf16)f0.x; o[1] = (__bf16)f0.y; o[2] = (__bf16)f0.z; o[3] = (__bf16)f0.w;
  o[4] = (__bf16)f1.x; o[5] = (__bf16)f1.y; o[6] = (__bf16)f1.z; o[7] = (__bf16)f1.w;
  ((v8bf*)xb)[t] = o;
}

// ---------------- prep 2: Wt[o][i*4+d] = wsum[o,i] * w[o,d] (bf16) ------------
__global__ __launch_bounds__(256) void build_w_kernel(const float* __restrict__ wsum,
                                                      const float* __restrict__ w,
                                                      __bf16* __restrict__ Wt) {
  int t = blockIdx.x * 256 + threadIdx.x;       // t = o*IDIM + i
  int o = t >> 10;
  int i = t & (IDIM - 1);
  float  s  = wsum[(size_t)o * IDIM + i];
  float4 wd = ((const float4*)w)[o];            // w[o][0..3]
  v4bf r;
  r[0] = (__bf16)(s * wd.x);
  r[1] = (__bf16)(s * wd.y);
  r[2] = (__bf16)(s * wd.z);
  r[3] = (__bf16)(s * wd.w);
  *(v4bf*)&Wt[(size_t)o * KDIM + i * 4] = r;
}

// ------------- main GEMM: 128x128x64 tiles, ping-pong LDS, WMMA bf16 ----------
__global__ __launch_bounds__(256) void spline_gemm_kernel(const __bf16* __restrict__ A,
                                                          const __bf16* __restrict__ B,
                                                          float* __restrict__ C) {
  __shared__ __bf16 sA[2][BM * LDT];
  __shared__ __bf16 sB[2][BN * LDT];

  const int tid  = threadIdx.x;
  const int lane = tid & 31;
  const int wave = tid >> 5;       // 0..7
  const int wM   = wave >> 1;      // 0..3 -> 32-row strip
  const int wN   = wave & 1;       // 0..1 -> 64-col strip
  const long bM  = (long)blockIdx.y * BM;
  const long bN  = (long)blockIdx.x * BN;

  // global->LDS: 16B chunks (8 bf16). BK=64 -> 8 chunks/row, 1024 chunks/tile,
  // 4 chunks per thread per matrix.
  const __bf16* gA[4];
  const __bf16* gB[4];
  int lofs[4];
#pragma unroll
  for (int j = 0; j < 4; ++j) {
    int c    = tid + j * 256;
    int row  = c >> 3;
    int koff = (c & 7) * 8;
    gA[j]   = A + (bM + row) * KDIM + koff;
    gB[j]   = B + (bN + row) * KDIM + koff;
    lofs[j] = row * LDT + koff;
  }

  // WMMA 16-bit A/B lane mapping (ISA 7.12.2): lane L<16 holds row/col L with
  // K = [0..7] and [16..23]; lane L+16 holds the same row/col with K = [8..15]
  // and [24..31].
  const int mrow  = lane & 15;
  const int khalf = (lane >> 4) * 8;

  v8f acc[2][4];
#pragma unroll
  for (int mt = 0; mt < 2; ++mt)
#pragma unroll
    for (int nt = 0; nt < 4; ++nt)
      acc[mt][nt] = (v8f)0.0f;

  // ---- prologue: stage K-tile 0 into buffer 0 ----
#if defined(USE_ASYNC_LDS)
#pragma unroll
  for (int j = 0; j < 4; ++j) {
    async_b128(gA[j], &sA[0][lofs[j]]);
    async_b128(gB[j], &sB[0][lofs[j]]);
  }
  wait_async0();
#else
  {
    v8bf ra[4], rb[4];
#pragma unroll
    for (int j = 0; j < 4; ++j) { ra[j] = *(const v8bf*)gA[j]; rb[j] = *(const v8bf*)gB[j]; }
#pragma unroll
    for (int j = 0; j < 4; ++j) {
      *(v8bf*)&sA[0][lofs[j]] = ra[j];
      *(v8bf*)&sB[0][lofs[j]] = rb[j];
    }
  }
#endif
  __syncthreads();

  int p = 0;
  for (int k = 0; k < KDIM; k += BK) {
    const int knext = (k + BK) & (KDIM - 1);   // wrap on last iter (discarded)

    // ---- stage next K-tile into buf p^1 (overlaps with WMMA below) ----
#if defined(USE_ASYNC_LDS)
#pragma unroll
    for (int j = 0; j < 4; ++j) {
      async_b128(gA[j] + knext, &sA[p ^ 1][lofs[j]]);
      async_b128(gB[j] + knext, &sB[p ^ 1][lofs[j]]);
    }
#else
    v8bf ra[4], rb[4];
#pragma unroll
    for (int j = 0; j < 4; ++j) {
      ra[j] = *(const v8bf*)(gA[j] + knext);
      rb[j] = *(const v8bf*)(gB[j] + knext);
    }
#endif

    // ---- compute on buf p: 2 K-substeps x (2x4) WMMA tiles ----
#pragma unroll
    for (int ks = 0; ks < 2; ++ks) {
      const int kb = ks * 32 + khalf;
      v16bf af[2], bfr[4];
#pragma unroll
      for (int mt = 0; mt < 2; ++mt) {
        const __bf16* q = &sA[p][(wM * 32 + mt * 16 + mrow) * LDT + kb];
        v8bf lo = *(const v8bf*)q;
        v8bf hi = *(const v8bf*)(q + 16);
        af[mt] = __builtin_shufflevector(lo, hi, 0, 1, 2, 3, 4, 5, 6, 7,
                                                 8, 9, 10, 11, 12, 13, 14, 15);
      }
#pragma unroll
      for (int nt = 0; nt < 4; ++nt) {
        const __bf16* q = &sB[p][(wN * 64 + nt * 16 + mrow) * LDT + kb];
        v8bf lo = *(const v8bf*)q;
        v8bf hi = *(const v8bf*)(q + 16);
        bfr[nt] = __builtin_shufflevector(lo, hi, 0, 1, 2, 3, 4, 5, 6, 7,
                                                  8, 9, 10, 11, 12, 13, 14, 15);
      }
#pragma unroll
      for (int mt = 0; mt < 2; ++mt)
#pragma unroll
        for (int nt = 0; nt < 4; ++nt)
          acc[mt][nt] = __builtin_amdgcn_wmma_f32_16x16x32_bf16(
              false, af[mt], false, bfr[nt], (short)0, acc[mt][nt], false, false);
    }

    // ---- publish next tile, single barrier per iteration ----
#if defined(USE_ASYNC_LDS)
    wait_async0();
#else
#pragma unroll
    for (int j = 0; j < 4; ++j) {
      *(v8bf*)&sA[p ^ 1][lofs[j]] = ra[j];
      *(v8bf*)&sB[p ^ 1][lofs[j]] = rb[j];
    }
#endif
    __syncthreads();
    p ^= 1;
  }

  // C/D layout (ISA 7.12.2): VGPR r, lanes 0-15 -> M=r, lanes 16-31 -> M=r+8,
  // N = lane%16.
  const long rbase = bM + wM * 32 + (lane >> 4) * 8;
  const long cbase = bN + wN * 64 + (lane & 15);
#pragma unroll
  for (int mt = 0; mt < 2; ++mt)
#pragma unroll
    for (int nt = 0; nt < 4; ++nt)
#pragma unroll
      for (int r = 0; r < 8; ++r)
        C[(rbase + mt * 16 + r) * NDIM + cbase + nt * 16] = acc[mt][nt][r];
}

extern "C" void kernel_launch(void* const* d_in, const int* in_sizes, int n_in,
                              void* d_out, int out_size, void* d_ws, size_t ws_size,
                              hipStream_t stream) {
  const float* x    = (const float*)d_in[0];  // (B,H,I,4) f32
  const float* w    = (const float*)d_in[1];  // (O,4) f32
  const float* wsum = (const float*)d_in[2];  // (O,I) f32
  float* out = (float*)d_out;                 // (B,H,O) f32

  __bf16* xb = (__bf16*)d_ws;                                        // 64 MiB
  __bf16* Wt = (__bf16*)((char*)d_ws + (size_t)MDIM * KDIM * 2);     //  8 MiB

  const int n8 = MDIM * KDIM / 8;
  cvt_x_kernel<<<(n8 + 255) / 256, 256, 0, stream>>>(x, xb, n8);
  build_w_kernel<<<(NDIM * IDIM) / 256, 256, 0, stream>>>(wsum, w, Wt);

  dim3 grid(NDIM / BN, MDIM / BM);  // (8, 64)
  spline_gemm_kernel<<<grid, 256, 0, stream>>>(xb, Wt, out);
}
